// MultiModalAttention_74921409511932
// MI455X (gfx1250) — matrix operand order repaired and checked by reading, not compiled, not run on hardware
//
#include <hip/hip_runtime.h>
#include <hip/hip_bf16.h>

// ---------------------------------------------------------------------------
// MultiModalAttention for MI455X (gfx1250), bf16 WMMA path, fp32 accumulate.
// B=4, S=1024, D=1024, H=16, DK=64, LAMBDA1=LAMBDA2=1.
// ---------------------------------------------------------------------------

typedef __attribute__((ext_vector_type(16))) __bf16 v16bf;
typedef __attribute__((ext_vector_type(8)))  __bf16 v8bf;
typedef __attribute__((ext_vector_type(8)))  float  v8f;
typedef __attribute__((ext_vector_type(4)))  float  v4f;

#define WMMA_BF16(a, b, c) \
  __builtin_amdgcn_wmma_f32_16x16x32_bf16(false, (a), false, (b), (short)0, (c), false, false)

constexpr int Bc  = 4;
constexpr int Sc  = 1024;
constexpr int Dc  = 1024;
constexpr int DKc = 64;
constexpr int Mrows = Bc * Sc;          // 4096

// ---- cross-lane max within 16-lane rows (VALU permlane, DS fallback) -------
__device__ __forceinline__ float xlane(float x, unsigned s0, unsigned s1, int mask) {
#if __has_builtin(__builtin_amdgcn_permlane16)
  unsigned u = __builtin_bit_cast(unsigned, x);
  u = __builtin_amdgcn_permlane16(u, u, s0, s1, false, false);
  (void)mask;
  return __builtin_bit_cast(float, u);
#else
  (void)s0; (void)s1;
  return __shfl_xor(x, mask, 32);
#endif
}
__device__ __forceinline__ float rowmax16(float x) {
  x = fmaxf(x, xlane(x, 0x67452301u, 0xEFCDAB89u, 1));   // xor 1
  x = fmaxf(x, xlane(x, 0x54761032u, 0xDCFE98BAu, 2));   // xor 2
  x = fmaxf(x, xlane(x, 0x32107654u, 0xBA98FEDCu, 4));   // xor 4
  x = fmaxf(x, xlane(x, 0xFEDCBA98u, 0x76543210u, 8));   // xor 8
  return x;
}

// ---------------------------------------------------------------------------
// GEMM: C[M,N] = A[M,K] @ W[N,K]^T + bias[N]
// 256 threads = 8 waves. Block tile 128(M) x 128(N); wave tile 32x64
// (2x4 WMMA accumulators -> 8 WMMAs per 32-deep K step).
// ---------------------------------------------------------------------------
template <typename AT, typename OT>
__global__ __launch_bounds__(256)
void gemm_wmma(const AT* __restrict__ A, const float* __restrict__ W,
               const float* __restrict__ bias, OT* __restrict__ C,
               int M, int N, int K)
{
  constexpr int LDA = 40;                       // padded LDS stride (bf16 elems)
  __shared__ __bf16 sA[128 * LDA];              // 10240 B
  __shared__ __bf16 sW[128 * LDA];              // 10240 B

  const int tid  = threadIdx.x;
  const int lane = tid & 31;
  const int wave = tid >> 5;
  const int mw   = (wave >> 1) * 32;            // wave M offset in block tile
  const int nw   = (wave & 1) * 64;             // wave N offset in block tile

  const int m0 = blockIdx.x * 128;
  const int n0 = blockIdx.y * 128;

  const int am = lane & 15;                     // fragment row / col within 16
  const int ah = lane >> 4;                     // lane-half selector

  v8f acc[2][4] = {};

  const int r  = tid >> 1;                      // staging row 0..127
  const int c0 = (tid & 1) * 16;                // staging col 0 or 16

  for (int k0 = 0; k0 < K; k0 += 32) {
    // Stage A tile 128x32 (fp -> bf16, 16B vector traffic)
    {
      const AT* ap = A + (size_t)(m0 + r) * K + k0 + c0;
#pragma unroll
      for (int j = 0; j < 2; ++j) {
        v8bf o;
#pragma unroll
        for (int i = 0; i < 8; ++i) o[i] = (__bf16)((float)ap[j * 8 + i]);
        *(v8bf*)(sA + r * LDA + c0 + j * 8) = o;
      }
    }
    // Stage W tile 128x32
    {
      const float* wp = W + (size_t)(n0 + r) * K + k0 + c0;
#pragma unroll
      for (int j = 0; j < 2; ++j) {
        const v4f f0 = *(const v4f*)(wp + j * 8);
        const v4f f1 = *(const v4f*)(wp + j * 8 + 4);
        v8bf o;
#pragma unroll
        for (int i = 0; i < 4; ++i) { o[i] = (__bf16)f0[i]; o[4 + i] = (__bf16)f1[i]; }
        *(v8bf*)(sW + r * LDA + c0 + j * 8) = o;
      }
    }
    __syncthreads();

    // A fragments: 16x32 bf16, lane m holds row m; K split {0..7,16..23}/{8..15,24..31}
    v16bf af[2];
#pragma unroll
    for (int i = 0; i < 2; ++i) {
      const __bf16* ap = &sA[(mw + i * 16 + am) * LDA];
#pragma unroll
      for (int t = 0; t < 8; ++t) {
        af[i][t]     = ap[ah * 8 + t];
        af[i][8 + t] = ap[16 + ah * 8 + t];
      }
    }
    // B fragments: 32x16 bf16, lane n holds col n; K split {0..7,16..23}/{8..15,24..31}
    v16bf bf2[4];
#pragma unroll
    for (int j = 0; j < 4; ++j) {
      const __bf16* bp = &sW[(nw + j * 16 + am) * LDA];
#pragma unroll
      for (int t = 0; t < 16; ++t) bf2[j][t] = bp[(t >> 3) * 16 + ah * 8 + (t & 7)];
    }
#pragma unroll
    for (int i = 0; i < 2; ++i)
#pragma unroll
      for (int j = 0; j < 4; ++j)
        acc[i][j] = WMMA_BF16(af[i], bf2[j], acc[i][j]);

    __syncthreads();
  }

  // Epilogue: C layout — VGPR v: rows v (lanes 0-15) / v+8 (lanes 16-31), col = lane&15
#pragma unroll
  for (int i = 0; i < 2; ++i) {
#pragma unroll
    for (int j = 0; j < 4; ++j) {
      const int col = n0 + nw + j * 16 + am;
      const float bv = bias[col];
#pragma unroll
      for (int v = 0; v < 8; ++v) {
        const int row = m0 + mw + i * 16 + ah * 8 + v;
        C[(size_t)row * N + col] = (OT)(acc[i][j][v] + bv);
      }
    }
  }
}

// ---------------------------------------------------------------------------
// One attention stream (flash, online softmax), 64 keys per iteration,
// K/V staged cooperatively in LDS. Row sums computed with a ones-matrix
// WMMA (no permlane sum chains); only the row max uses permlane.
// ---------------------------------------------------------------------------
constexpr int KB  = 64;   // keys per block iteration
constexpr int LDK = 72;   // sK row stride  (64 keys x 64 dk)
constexpr int LDV = 72;   // sVT row stride (64 dk x 64 keys)
constexpr int LP  = 72;   // per-wave P-tile stride (16 q x 64 keys)

__device__ __forceinline__ void stream_flash(
    const __bf16* __restrict__ Kp, const __bf16* __restrict__ Vp,
    const v16bf (&qa)[2], const v16bf& vones,
    __bf16* sK, __bf16* sVT, __bf16* myP,
    v8f (&total)[4], int tid, int am, int ah)
{
  v8f O[4] = {};
  float m_i[8], l_i[8];
#pragma unroll
  for (int v = 0; v < 8; ++v) { m_i[v] = -3.0e38f; l_i[v] = 0.0f; }

  const float scale = 0.125f;                   // 1/sqrt(DK)

  for (int kb = 0; kb < Sc; kb += KB) {
    // ---- cooperative staging: K (64x64 row-major) and V^T (64x64) ----
    {
      const int r  = tid >> 2;                  // 0..63 (key row)
      const int c0 = (tid & 3) * 16;            // 0,16,32,48 (dk col)
#pragma unroll
      for (int j = 0; j < 2; ++j) {
        const v8bf kv = *(const v8bf*)(Kp + (size_t)(kb + r) * Dc + c0 + j * 8);
        *(v8bf*)(sK + r * LDK + c0 + j * 8) = kv;
        const v8bf vv = *(const v8bf*)(Vp + (size_t)(kb + r) * Dc + c0 + j * 8);
#pragma unroll
        for (int i = 0; i < 8; ++i) sVT[(c0 + j * 8 + i) * LDV + r] = vv[i];
      }
    }
    __syncthreads();

    // ---- scores: 4 key tiles x 2 dk halves = 8 WMMAs ----
    v8f c[4] = {};
#pragma unroll
    for (int half = 0; half < 2; ++half) {
#pragma unroll
      for (int kt = 0; kt < 4; ++kt) {
        v16bf b;
        const __bf16* kr = sK + (kt * 16 + am) * LDK + half * 32;
#pragma unroll
        for (int t = 0; t < 16; ++t) b[t] = kr[(t >> 3) * 16 + ah * 8 + (t & 7)];
        c[kt] = WMMA_BF16(qa[half], b, c[kt]);
      }
    }

    // ---- online softmax over 64 keys (one reduction chain per row) ----
    float alpha[8];
#pragma unroll
    for (int v = 0; v < 8; ++v) {
      const float s0 = c[0][v] * scale;
      const float s1 = c[1][v] * scale;
      const float s2 = c[2][v] * scale;
      const float s3 = c[3][v] * scale;
      const float mx = rowmax16(fmaxf(fmaxf(s0, s1), fmaxf(s2, s3)));
      const float nm = fmaxf(m_i[v], mx);
      alpha[v] = __expf(m_i[v] - nm);
      m_i[v] = nm;
      const float p0 = __expf(s0 - nm);
      const float p1 = __expf(s1 - nm);
      const float p2 = __expf(s2 - nm);
      const float p3 = __expf(s3 - nm);
      const int r = v + ah * 8;                 // C-layout row -> LDS row-major
      myP[r * LP + am]      = (__bf16)p0;
      myP[r * LP + 16 + am] = (__bf16)p1;
      myP[r * LP + 32 + am] = (__bf16)p2;
      myP[r * LP + 48 + am] = (__bf16)p3;
    }
#pragma unroll
    for (int t = 0; t < 4; ++t)
#pragma unroll
      for (int v = 0; v < 8; ++v) O[t][v] *= alpha[v];

    asm volatile("s_wait_dscnt 0" ::: "memory");  // per-wave P write->read

    // ---- re-read P as two A fragments (keys 0..31 and 32..63) ----
    v16bf pa0, pa1;
    {
      const __bf16* pr = &myP[am * LP];
#pragma unroll
      for (int t = 0; t < 8; ++t) {
        pa0[t]     = pr[ah * 8 + t];
        pa0[8 + t] = pr[16 + ah * 8 + t];
        pa1[t]     = pr[32 + ah * 8 + t];
        pa1[8 + t] = pr[48 + ah * 8 + t];
      }
    }

    // ---- row sums via ones-matmul: every lane gets its rows' sums ----
    v8f ls = {};
    ls = WMMA_BF16(pa0, vones, ls);
    ls = WMMA_BF16(pa1, vones, ls);
#pragma unroll
    for (int v = 0; v < 8; ++v) l_i[v] = l_i[v] * alpha[v] + ls[v];

    // ---- O += P @ V  (V^T staged: B-frag per lane is two contiguous 16B runs) ----
#pragma unroll
    for (int t4 = 0; t4 < 4; ++t4) {
      v16bf vb0, vb1;
      const __bf16* vr = sVT + (t4 * 16 + am) * LDV;
#pragma unroll
      for (int u = 0; u < 16; ++u) { const int o = (u >> 3) * 16 + ah * 8 + (u & 7); vb0[u] = vr[o]; vb1[u] = vr[32 + o]; }
      O[t4] = WMMA_BF16(pa0, vb0, O[t4]);
      O[t4] = WMMA_BF16(pa1, vb1, O[t4]);
    }
    __syncthreads();                            // protect sK/sVT for next block
  }

  // normalize this stream and add into running sum
#pragma unroll
  for (int v = 0; v < 8; ++v) {
    const float inv = 1.0f / l_i[v];
#pragma unroll
    for (int t = 0; t < 4; ++t) total[t][v] += O[t][v] * inv;
  }
}

// ---------------------------------------------------------------------------
// Fused 4-stream attention: one block = one (b,h) x 128 query rows, 8 waves
// of 16 q-rows each. Streams e/i/ei/ie accumulated in registers (l1=l2=1).
// ---------------------------------------------------------------------------
__global__ __launch_bounds__(256)
void attn_wmma(const __bf16* __restrict__ qp,
               const __bf16* __restrict__ k0p, const __bf16* __restrict__ v0p,
               const __bf16* __restrict__ k1p, const __bf16* __restrict__ v1p,
               const __bf16* __restrict__ k2p, const __bf16* __restrict__ v2p,
               const __bf16* __restrict__ k3p, const __bf16* __restrict__ v3p,
               __bf16* __restrict__ outp)
{
  __shared__ __bf16 sK [64 * LDK];              //  9216 B
  __shared__ __bf16 sVT[64 * LDV];              //  9216 B
  __shared__ __bf16 sP [8 * 16 * LP];           // 18432 B

  const int tid  = threadIdx.x;
  const int lane = tid & 31;
  const int wave = tid >> 5;

  const int qblk = blockIdx.x & 7;              // 8 q-blocks of 128 rows
  const int bh   = blockIdx.x >> 3;             // 0..63
  const int b    = bh >> 4;
  const int h    = bh & 15;

  const int q0 = qblk * 128 + wave * 16;
  const size_t base_q  = ((size_t)(b * Sc + q0)) * Dc + h * DKc;
  const size_t base_kv = ((size_t)(b * Sc)) * Dc + h * DKc;

  const int am = lane & 15;
  const int ah = lane >> 4;

  // Q fragments (A-matrix 16x32 per 32-wide dk slice), resident all kernel
  v16bf qa[2];
#pragma unroll
  for (int kk = 0; kk < 2; ++kk) {
    const __bf16* qrow = qp + base_q + (size_t)am * Dc + kk * 32;
#pragma unroll
    for (int t = 0; t < 8; ++t) {
      qa[kk][t]     = qrow[ah * 8 + t];
      qa[kk][8 + t] = qrow[16 + ah * 8 + t];
    }
  }

  v16bf vones;
#pragma unroll
  for (int t = 0; t < 16; ++t) vones[t] = (__bf16)1.0f;

  __bf16* myP = &sP[wave * 16 * LP];
  v8f total[4] = {};

  stream_flash(k0p + base_kv, v0p + base_kv, qa, vones, sK, sVT, myP, total, tid, am, ah);
  stream_flash(k1p + base_kv, v1p + base_kv, qa, vones, sK, sVT, myP, total, tid, am, ah);
  stream_flash(k2p + base_kv, v2p + base_kv, qa, vones, sK, sVT, myP, total, tid, am, ah);
  stream_flash(k3p + base_kv, v3p + base_kv, qa, vones, sK, sVT, myP, total, tid, am, ah);

  // store summed heads in [B*S, D] layout (head-concat is contiguous in D)
#pragma unroll
  for (int t = 0; t < 4; ++t)
#pragma unroll
    for (int v = 0; v < 8; ++v) {
      const int row = q0 + ah * 8 + v;
      const int col = h * DKc + t * 16 + am;
      outp[((size_t)(b * Sc + row)) * Dc + col] = (__bf16)total[t][v];
    }
}

// ---------------------------------------------------------------------------
// Launch: 9 projection GEMMs -> fused 4-stream attention -> output GEMM.
// Workspace: 10 x (4096*1024) bf16 buffers = 80 MB.
// ---------------------------------------------------------------------------
extern "C" void kernel_launch(void* const* d_in, const int* in_sizes, int n_in,
                              void* d_out, int out_size, void* d_ws, size_t ws_size,
                              hipStream_t stream)
{
  (void)in_sizes; (void)n_in; (void)out_size; (void)ws_size;

  char* ws = (char*)d_ws;
  const size_t projBytes = (size_t)Mrows * Dc * sizeof(__bf16);   // 8 MB
  __bf16* buf[10];
  for (int i = 0; i < 10; ++i) buf[i] = (__bf16*)(ws + (size_t)i * projBytes);
  // buf: 0=Q, 1=Ke, 2=Ve, 3=Ki, 4=Vi, 5=Kei, 6=Vei, 7=Kie, 8=Vie, 9=att-out

  // input index map: activations 0..8, weights 9..18, biases 19..28
  struct PJ { int xi, wi, bi; };
  const PJ pj[9] = {
    {0,  9, 19},   // Q   = q    @ Wq^T   + bq
    {2, 10, 20},   // Ke  = k_e  @ Wke^T  + bke
    {1, 11, 21},   // Ve  = v_e  @ Wve^T  + bve
    {4, 12, 22},   // Ki  = k_i  @ Wki^T  + bki
    {3, 13, 23},   // Vi  = v_i  @ Wvi^T  + bvi
    {6, 14, 24},   // Kei = k_ei @ Wkei^T + bkei
    {5, 15, 25},   // Vei = v_ei @ Wvei^T + bvei
    {8, 16, 26},   // Kie = k_ie @ Wkie^T + bkie
    {7, 17, 27},   // Vie = v_ie @ Wvie^T + bvie
  };

  dim3 gg(Mrows / 128, Dc / 128);  // 32 x 8
  dim3 bb(256);

  for (int i = 0; i < 9; ++i) {
    gemm_wmma<float, __bf16><<<gg, bb, 0, stream>>>(
        (const float*)d_in[pj[i].xi], (const float*)d_in[pj[i].wi],
        (const float*)d_in[pj[i].bi], buf[i], Mrows, Dc, Dc);
  }

  // attention: 512 blocks = (B*H=64) x (S/128=8)
  attn_wmma<<<dim3(512), bb, 0, stream>>>(
      buf[0],
      buf[1], buf[2],
      buf[3], buf[4],
      buf[5], buf[6],
      buf[7], buf[8],
      buf[9]);

  // out = att @ Wout^T + bout  (fp32 result)
  gemm_wmma<__bf16, float><<<gg, bb, 0, stream>>>(
      buf[9], (const float*)d_in[18], (const float*)d_in[28],
      (float*)d_out, Mrows, Dc, Dc);
}
